// GRU_28939489640938
// MI455X (gfx1250) — compile-verified, matrix operand order checked by reading, not run on hardware
//
#include <hip/hip_runtime.h>

#define I_SZ 128
#define H_SZ 128
#define N_ENV 256
#define T_STEPS 2048
#define ENV_TILE 16
#define NTHREADS 512

typedef __attribute__((ext_vector_type(16))) __bf16 v16bf;
typedef __attribute__((ext_vector_type(8)))  float  v8f;

// Native f32->bf16 (RNE) — let the backend use the hardware cvt path.
__device__ __forceinline__ __bf16 f2bf(float f) { return (__bf16)f; }

// K index for element e of a 16-bit A/B WMMA fragment (16x32 / 32x16):
// half 0 holds K = kk+[0..7] and kk+[16..23]; half 1 the +8 groups.
__device__ __forceinline__ int kmap(int kk, int lh, int e) {
  return kk + 8 * lh + e + ((e >= 8) ? 8 : 0);
}

// B[k][n] = w[g][k] with g = column (gbase + lane%16): B = W^T, loaded once into VGPRs.
__device__ __forceinline__ v16bf load_b_frag(const float* __restrict__ w, int g, int kk, int lh) {
  v16bf b;
#pragma unroll
  for (int e = 0; e < 16; ++e) b[e] = f2bf(w[g * 128 + kmap(kk, lh, e)]);
  return b;
}

__device__ __forceinline__ v16bf load_a_x(const float* __restrict__ xrow, int kk, int lh) {
  v16bf a;
#pragma unroll
  for (int e = 0; e < 16; ++e) a[e] = f2bf(xrow[kmap(kk, lh, e)]);
  return a;
}

__device__ __forceinline__ v16bf load_a_h(const float* __restrict__ hrow, float mm, int kk, int lh) {
  v16bf a;
#pragma unroll
  for (int e = 0; e < 16; ++e) a[e] = f2bf(hrow[kmap(kk, lh, e)] * mm);
  return a;
}

__global__ __launch_bounds__(NTHREADS, 1)
void gru_wmma_scan_28939489640938(const float* __restrict__ x,
                                  const float* __restrict__ h0,
                                  const float* __restrict__ masks,
                                  const float* __restrict__ w_ih,
                                  const float* __restrict__ w_hh,
                                  const float* __restrict__ b_ih,
                                  const float* __restrict__ b_hh,
                                  float* __restrict__ out) {
  __shared__ float h_lds[ENV_TILE][H_SZ];        // resident hidden state tile (fp32)
  __shared__ float pre_rz[ENV_TILE][2 * H_SZ];   // fused r/z preactivations
  __shared__ float pre_in[ENV_TILE][H_SZ];       // x-side n-gate preactivation
  __shared__ float pre_hn[ENV_TILE][H_SZ];       // h-side n-gate preactivation
  __shared__ float mask_lds[ENV_TILE];

  const int tid  = threadIdx.x;
  const int wave = tid >> 5;     // 0..15 (wave32)
  const int lane = tid & 31;
  const int ln   = lane & 15;    // M (A) / N (B,C,D) index
  const int lh   = lane >> 4;    // half-wave selector
  const int e0   = blockIdx.x * ENV_TILE;

  // Fused r/z tile owned by this wave: columns g1 in [16w, 16w+16) of [0,256)
  const int g1 = 16 * wave + ln;
  // Second tile: n-gate columns in [256,384). Waves 0-7 -> i_n (x side), 8-15 -> h_n (h side)
  // Force a *scalar* branch condition so EXEC is never masked around WMMA.
  const int second_x = __builtin_amdgcn_readfirstlane((wave < 8) ? 1 : 0);
  const int g2 = 256 + 16 * (wave & 7) + ln;

  const float bias1 = b_ih[g1] + b_hh[g1];
  const float bias2 = second_x ? b_ih[g2] : b_hh[g2];

  // ---- One-time: weight B-fragments into registers (bf16) ----
  v16bf bw1[8];
#pragma unroll
  for (int kb = 0; kb < 4; ++kb) bw1[kb]     = load_b_frag(w_ih, g1, 32 * kb, lh);
#pragma unroll
  for (int kb = 0; kb < 4; ++kb) bw1[kb + 4] = load_b_frag(w_hh, g1, 32 * kb, lh);
  v16bf bw2[4];
  {
    const float* w2 = second_x ? w_ih : w_hh;
#pragma unroll
    for (int kb = 0; kb < 4; ++kb) bw2[kb] = load_b_frag(w2, g2, 32 * kb, lh);
  }

  // ---- Load initial hidden state ----
  for (int idx = tid; idx < ENV_TILE * H_SZ; idx += NTHREADS) {
    int m = idx >> 7, j = idx & 127;
    h_lds[m][j] = h0[(size_t)(e0 + m) * H_SZ + j];
  }
  __syncthreads();

  for (int t = 0; t < T_STEPS; ++t) {
    if (tid < ENV_TILE) mask_lds[tid] = masks[(size_t)t * N_ENV + e0 + tid];
    __syncthreads();  // mask + h ready

    const float* xrow = x + ((size_t)t * N_ENV + e0 + ln) * I_SZ;  // row m = ln
    const float  mm   = mask_lds[ln];

    v8f c1, c2;
#pragma unroll
    for (int r = 0; r < 8; ++r) { c1[r] = bias1; c2[r] = bias2; }

    // x-side K chunks: build each A fragment once, consume immediately in both tiles.
#pragma unroll
    for (int kb = 0; kb < 4; ++kb) {
      v16bf a = load_a_x(xrow, 32 * kb, lh);
      c1 = __builtin_amdgcn_wmma_f32_16x16x32_bf16(false, a, false, bw1[kb], (short)0, c1, false, false);
      if (second_x)
        c2 = __builtin_amdgcn_wmma_f32_16x16x32_bf16(false, a, false, bw2[kb], (short)0, c2, false, false);
    }
    // h-side K chunks (masked hidden state from LDS).
#pragma unroll
    for (int kb = 0; kb < 4; ++kb) {
      v16bf a = load_a_h(&h_lds[ln][0], mm, 32 * kb, lh);
      c1 = __builtin_amdgcn_wmma_f32_16x16x32_bf16(false, a, false, bw1[kb + 4], (short)0, c1, false, false);
      if (!second_x)
        c2 = __builtin_amdgcn_wmma_f32_16x16x32_bf16(false, a, false, bw2[kb], (short)0, c2, false, false);
    }

    // D layout: element r of lane l -> row (r + 8*(l/16)), col (l%16)
#pragma unroll
    for (int r = 0; r < 8; ++r) {
      int row = r + 8 * lh;
      pre_rz[row][g1] = c1[r];
      if (second_x) pre_in[row][g2 - 256] = c2[r];
      else          pre_hn[row][g2 - 256] = c2[r];
    }
    __syncthreads();  // preactivations ready

    // Elementwise gates in fp32; update resident h and stream out h_new
    const size_t obase = ((size_t)t * N_ENV + e0) * H_SZ;
    for (int idx = tid; idx < ENV_TILE * H_SZ; idx += NTHREADS) {
      int m = idx >> 7, j = idx & 127;
      float r = pre_rz[m][j];
      float z = pre_rz[m][j + H_SZ];
      r = 1.0f / (1.0f + __expf(-r));
      z = 1.0f / (1.0f + __expf(-z));
      float n    = tanhf(pre_in[m][j] + r * pre_hn[m][j]);
      float hm   = h_lds[m][j] * mask_lds[m];
      float hnew = (1.0f - z) * n + z * hm;
      h_lds[m][j] = hnew;
      out[obase + (size_t)m * H_SZ + j] = hnew;
    }

    // Prefetch next step's x tile (gfx1250 global_prefetch path)
    if (t + 1 < T_STEPS) {
      const float* xn = x + ((size_t)(t + 1) * N_ENV + e0 + ln) * I_SZ + 64 * lh;
      __builtin_prefetch(xn, 0, 3);
    }
    __syncthreads();  // h updated before next step's A-fragment reads
  }

  // h_final appended after outs
  for (int idx = tid; idx < ENV_TILE * H_SZ; idx += NTHREADS) {
    int m = idx >> 7, j = idx & 127;
    out[(size_t)T_STEPS * N_ENV * H_SZ + (size_t)(e0 + m) * H_SZ + j] = h_lds[m][j];
  }
}

extern "C" void kernel_launch(void* const* d_in, const int* in_sizes, int n_in,
                              void* d_out, int out_size, void* d_ws, size_t ws_size,
                              hipStream_t stream) {
  (void)in_sizes; (void)n_in; (void)d_ws; (void)ws_size; (void)out_size;
  const float* x     = (const float*)d_in[0];
  const float* h0    = (const float*)d_in[1];
  const float* masks = (const float*)d_in[2];
  const float* w_ih  = (const float*)d_in[3];
  const float* w_hh  = (const float*)d_in[4];
  const float* b_ih  = (const float*)d_in[5];
  const float* b_hh  = (const float*)d_in[6];
  float* out = (float*)d_out;

  dim3 grid(N_ENV / ENV_TILE);  // 16 independent env-tiles, no cross-WG sync
  dim3 block(NTHREADS);         // 16 wave32 waves
  gru_wmma_scan_28939489640938<<<grid, block, 0, stream>>>(
      x, h0, masks, w_ih, w_hh, b_ih, b_hh, out);
}
